// LIIF_26156350833318
// MI455X (gfx1250) — matrix operand order repaired and checked by reading, not compiled, not run on hardware
//
#include <hip/hip_runtime.h>
#include <hip/hip_bf16.h>

typedef __attribute__((ext_vector_type(16))) _Float16 v16h;
typedef __attribute__((ext_vector_type(8)))  float    v8f;

// ---------------- problem constants ----------------
#define BQ      65536      // B*Q
#define QMASK   32767      // Q-1 (Q=32768)
#define QSHIFT  15
#define HDIM    128
#define WDIM    128
#define HW      16384
#define CDIM    64
#define HID     256
#define NT      16         // N tiles of 16 in hidden dim
#define NHALF   8          // N tiles per half
// fragment counts per layer (16x16x32 f16 tiles of B matrix)
#define L0_FRAGS  48       // 3 ktiles (K padded 68->96) x 16 ntiles
#define LH_FRAGS  128      // 8 ktiles x 16 ntiles
#define L4_FRAGS  8        // 8 ktiles x 1 ntile (N padded 3->16)
#define TOT_FRAGS (L0_FRAGS + 3*LH_FRAGS + L4_FRAGS)   // 440
// workspace offsets in halves (each frag = 512 halves = 1KB)
#define WF_L0   0
#define WF_L1   (48*512)
#define WF_L2   (176*512)
#define WF_L3   (304*512)
#define WF_L4   (432*512)
// LDS layout (bytes)
#define WBUF_OFF  0u
#define WBUF_MAX  131072u                // 128 KB (largest layer)
#define ACT0_OFF  131072u                // 8 waves * 16 rows * 256 halves * 2B
#define ACT1_OFF  196608u                // ping-pong partner
#define ACT_WSTRIDE 8192u
#define AREA_OFF  262144u                // 128 f32
#define PIX_OFF   262656u                // 128 u32
#define OUT_OFF   263168u                // 8 waves * 16 rows * 4 f32
#define SMEM_BYTES 265216u

union Frag16 { v16h h; float4 f[2]; };

__device__ __forceinline__ v8f zero8() {
  v8f z;
#pragma unroll
  for (int i = 0; i < 8; ++i) z[i] = 0.0f;
  return z;
}

// ---- async global -> LDS weight staging (CDNA5 path, ASYNCcnt) ----
__device__ __forceinline__ void issue_async_weights(char* smem, const _Float16* src,
                                                    int bytes, int tid) {
  uint64_t sbase = (uint64_t)(uintptr_t)src;
  uint32_t lbase = (uint32_t)(uintptr_t)smem;   // WBUF at offset 0 of dyn-LDS
  for (int i = tid * 16; i < bytes; i += 256 * 16) {
    uint32_t lds  = lbase + (uint32_t)i;
    uint32_t voff = (uint32_t)i;
    asm volatile("global_load_async_to_lds_b128 %0, %1, %2"
                 :: "v"(lds), "v"(voff), "s"(sbase) : "memory");
  }
}
__device__ __forceinline__ void wait_async() {
  asm volatile("s_wait_asynccnt 0" ::: "memory");
}

// ---- GEMM for one N-half of a layer: acc[j] += A(16xK) * Wfrag ----
// Double-buffered k-pipeline: while 8 WMMAs consume k-tile kt, the 16
// ds_load_b128 of k-tile kt+1 are in flight (8-WMMA load->use distance).
template<int KT>
__device__ __forceinline__ void gemm_half(v8f acc[NHALF], const char* smem,
                                          uint32_t actb, int lane, int half) {
  const int m    = lane & 15;
  const int asel = (lane >> 4) * 8;   // A: halves 0-7 -> K=asel.., 8-15 -> K=16+asel..
  const uint32_t abase = actb + (uint32_t)(m * 512 + asel * 2);
  const uint32_t bbase = (uint32_t)(lane * 32) + (uint32_t)(half * NHALF * 1024);

  Frag16 a[2], bg[2][NHALF];
  a[0].f[0] = *(const float4*)(smem + abase);
  a[0].f[1] = *(const float4*)(smem + abase + 32);
#pragma unroll
  for (int j = 0; j < NHALF; ++j) {
    const uint32_t off = bbase + (uint32_t)j * 1024;
    bg[0][j].f[0] = *(const float4*)(smem + off);
    bg[0][j].f[1] = *(const float4*)(smem + off + 16);
  }

#pragma unroll
  for (int kt = 0; kt < KT; ++kt) {
    if (kt + 1 < KT) {
      const uint32_t aoff = abase + (uint32_t)(kt + 1) * 64;
      a[(kt + 1) & 1].f[0] = *(const float4*)(smem + aoff);
      a[(kt + 1) & 1].f[1] = *(const float4*)(smem + aoff + 32);
      const uint32_t gb = bbase + (uint32_t)((kt + 1) * NT * 1024);
#pragma unroll
      for (int j = 0; j < NHALF; ++j) {
        bg[(kt + 1) & 1][j].f[0] = *(const float4*)(smem + gb + (uint32_t)j * 1024);
        bg[(kt + 1) & 1][j].f[1] = *(const float4*)(smem + gb + (uint32_t)j * 1024 + 16);
      }
    }
#pragma unroll
    for (int j = 0; j < NHALF; ++j) {
      acc[j] = __builtin_amdgcn_wmma_f32_16x16x32_f16(
          false, a[kt & 1].h, false, bg[kt & 1][j].h, (short)0, acc[j], false, false);
    }
  }
}

// ---- bias + relu + f16 store of one N-half of D tiles into act_out ----
__device__ __forceinline__ void epilogue_half(v8f acc[NHALF], const float* bias,
                                              char* smem, uint32_t actb_out,
                                              int lane, int half) {
  const int col = lane & 15;
  const int rb  = (lane >> 4) * 8;    // D: vgpr j -> row rb+j, col = lane&15
#pragma unroll
  for (int n = 0; n < NHALF; ++n) {
    const int nt = half * NHALF + n;
    float bv = bias[nt * 16 + col];
#pragma unroll
    for (int j = 0; j < 8; ++j) {
      float v = acc[n][j] + bv;
      v = fmaxf(v, 0.0f);
      *(_Float16*)(smem + actb_out +
                   (uint32_t)((rb + j) * 512 + (nt * 16 + col) * 2)) = (_Float16)v;
    }
  }
}

// ================= weight pre-swizzle: f32 -> f16 B-fragments =================
__global__ __launch_bounds__(256) void liif_prep_kernel(
    const float* __restrict__ w0, const float* __restrict__ w1,
    const float* __restrict__ w2, const float* __restrict__ w3,
    const float* __restrict__ w4, _Float16* __restrict__ wf) {
  int t = blockIdx.x * 256 + threadIdx.x;
  int frag = t >> 5, lane = t & 31;
  if (frag >= TOT_FRAGS) return;
  const int n_lo = lane & 15;
  const int kb   = (lane >> 4) * 8;           // B: halves 0-7 -> K=kb.., 8-15 -> K=16+kb..
  int layer, kt, nt;
  const float* W;
  if (frag < 48)        { layer = 0; kt = frag / 16;        nt = frag % 16;        W = w0; }
  else if (frag < 176)  { layer = 1; kt = (frag-48) / 16;   nt = (frag-48) % 16;   W = w1; }
  else if (frag < 304)  { layer = 2; kt = (frag-176) / 16;  nt = (frag-176) % 16;  W = w2; }
  else if (frag < 432)  { layer = 3; kt = (frag-304) / 16;  nt = (frag-304) % 16;  W = w3; }
  else                  { layer = 4; kt = frag - 432;       nt = 0;                W = w4; }
  Frag16 o;
#pragma unroll
  for (int h = 0; h < 16; ++h) {
    int k = kt * 32 + kb + (h & 7) + ((h >> 3) << 4);
    float v;
    if (layer == 0)       v = (k < 68) ? w0[k * HID + nt * 16 + n_lo] : 0.0f;
    else if (layer == 4)  v = (n_lo < 3) ? w4[k * 3 + n_lo] : 0.0f;
    else                  v = W[k * HID + nt * 16 + n_lo];
    o.h[h] = (_Float16)v;
  }
  float4* dst = (float4*)((char*)wf + (size_t)frag * 1024 + (size_t)lane * 32);
  dst[0] = o.f[0];
  dst[1] = o.f[1];
}

// ================= main fused kernel =================
__global__ __launch_bounds__(256) void liif_main_kernel(
    const float* __restrict__ feat, const float* __restrict__ grid,
    const float* __restrict__ cell,
    const float* __restrict__ b0, const float* __restrict__ b1,
    const float* __restrict__ b2, const float* __restrict__ b3,
    const float* __restrict__ b4,
    const _Float16* __restrict__ wf, float* __restrict__ out) {
  extern __shared__ char smem[];
  const int tid  = threadIdx.x;
  const int wv   = tid >> 5;
  const int lane = tid & 31;
  const int base_row = (blockIdx.x * 8 + wv) * 16;   // 16 rows = 4 queries x 4 ensembles
  const uint32_t act0 = ACT0_OFF + (uint32_t)wv * ACT_WSTRIDE;
  const uint32_t act1 = ACT1_OFF + (uint32_t)wv * ACT_WSTRIDE;

  // kick off layer-0 weight staging, overlap with input build
  issue_async_weights(smem, wf + WF_L0, L0_FRAGS * 1024, tid);

  // ---- per-row scalar setup (lanes 0..15, lane == local row) ----
  if (lane < 16) {
    const int rg = base_row + lane;
    const int bq = rg >> 2;
    const int e  = rg & 3;                       // e = r_idx*2 + c_idx
    const int b  = bq >> QSHIFT;
    const float gy = grid[(size_t)bq * 2 + 0];
    const float gx = grid[(size_t)bq * 2 + 1];
    const float rs = (e & 2) ? 1.0f : -1.0f;
    const float cs = (e & 1) ? 1.0f : -1.0f;
    const float eps = 1e-6f;
    float fg0 = fminf(fmaxf(gy + rs * (1.0f / HDIM), -1.0f + eps), 1.0f - eps);
    float fg1 = fminf(fmaxf(gx + cs * (1.0f / WDIM), -1.0f + eps), 1.0f - eps);
    float iyf = fminf(fmaxf(rintf(((fg0 + 1.0f) * HDIM - 1.0f) * 0.5f), 0.0f), HDIM - 1.0f);
    float ixf = fminf(fmaxf(rintf(((fg1 + 1.0f) * WDIM - 1.0f) * 0.5f), 0.0f), WDIM - 1.0f);
    int iy = (int)iyf, ix = (int)ixf;
    uint32_t foff = (uint32_t)(b * CDIM * HW + iy * WDIM + ix);
    float coord0 = -1.0f + (2.0f * iyf + 1.0f) / HDIM;
    float coord1 = -1.0f + (2.0f * ixf + 1.0f) / WDIM;
    float qg0 = (gy - coord0) * HDIM;
    float qg1 = (gx - coord1) * WDIM;
    float qc0 = cell[(size_t)bq * 2 + 0] * HDIM;
    float qc1 = cell[(size_t)bq * 2 + 1] * WDIM;
    *(float*)(smem + AREA_OFF + (uint32_t)(wv * 16 + lane) * 4) =
        fabsf(qg0 * qg1) + 1e-9f;
    *(uint32_t*)(smem + PIX_OFF + (uint32_t)(wv * 16 + lane) * 4) = foff;
    _Float16* arow = (_Float16*)(smem + act0 + (uint32_t)lane * 512);
    arow[64] = (_Float16)qg0;
    arow[65] = (_Float16)qg1;
    arow[66] = (_Float16)qc0;
    arow[67] = (_Float16)qc1;
#pragma unroll
    for (int z = 68; z < 96; z += 4)             // zero K-padding (68..95), 8B stores
      *(uint2*)(smem + act0 + (uint32_t)lane * 512 + (uint32_t)z * 2) = make_uint2(0u, 0u);
  }

  // ---- feature gather: lane = channel pair, loop rows ----
  for (int r = 0; r < 16; ++r) {
    uint32_t fo = *(const uint32_t*)(smem + PIX_OFF + (uint32_t)(wv * 16 + r) * 4);
    float f0 = feat[(size_t)fo + (size_t)lane * HW];
    float f1 = feat[(size_t)fo + (size_t)(lane + 32) * HW];
    _Float16* arow = (_Float16*)(smem + act0 + (uint32_t)r * 512);
    arow[lane]      = (_Float16)f0;
    arow[lane + 32] = (_Float16)f1;
  }

  wait_async();
  __syncthreads();

  v8f acc[NHALF];

  // ================= layer 0: K=96 (3 ktiles), act0 -> act1 =================
#pragma unroll
  for (int j = 0; j < NHALF; ++j) acc[j] = zero8();
  gemm_half<3>(acc, smem, act0, lane, 0);
  epilogue_half(acc, b0, smem, act1, lane, 0);
#pragma unroll
  for (int j = 0; j < NHALF; ++j) acc[j] = zero8();
  gemm_half<3>(acc, smem, act0, lane, 1);
  __syncthreads();                                   // all waves done with w0
  issue_async_weights(smem, wf + WF_L1, LH_FRAGS * 1024, tid);
  epilogue_half(acc, b0, smem, act1, lane, 1);       // overlaps async copy
  wait_async();
  __syncthreads();

  // ================= layer 1: act1 -> act0 =================
#pragma unroll
  for (int j = 0; j < NHALF; ++j) acc[j] = zero8();
  gemm_half<8>(acc, smem, act1, lane, 0);
  epilogue_half(acc, b1, smem, act0, lane, 0);
#pragma unroll
  for (int j = 0; j < NHALF; ++j) acc[j] = zero8();
  gemm_half<8>(acc, smem, act1, lane, 1);
  __syncthreads();
  issue_async_weights(smem, wf + WF_L2, LH_FRAGS * 1024, tid);
  epilogue_half(acc, b1, smem, act0, lane, 1);
  wait_async();
  __syncthreads();

  // ================= layer 2: act0 -> act1 =================
#pragma unroll
  for (int j = 0; j < NHALF; ++j) acc[j] = zero8();
  gemm_half<8>(acc, smem, act0, lane, 0);
  epilogue_half(acc, b2, smem, act1, lane, 0);
#pragma unroll
  for (int j = 0; j < NHALF; ++j) acc[j] = zero8();
  gemm_half<8>(acc, smem, act0, lane, 1);
  __syncthreads();
  issue_async_weights(smem, wf + WF_L3, LH_FRAGS * 1024, tid);
  epilogue_half(acc, b2, smem, act1, lane, 1);
  wait_async();
  __syncthreads();

  // ================= layer 3: act1 -> act0 =================
#pragma unroll
  for (int j = 0; j < NHALF; ++j) acc[j] = zero8();
  gemm_half<8>(acc, smem, act1, lane, 0);
  epilogue_half(acc, b3, smem, act0, lane, 0);
#pragma unroll
  for (int j = 0; j < NHALF; ++j) acc[j] = zero8();
  gemm_half<8>(acc, smem, act1, lane, 1);
  __syncthreads();
  issue_async_weights(smem, wf + WF_L4, L4_FRAGS * 1024, tid);
  epilogue_half(acc, b3, smem, act0, lane, 1);
  wait_async();
  __syncthreads();

  // ========= layer 4: K=256 (8 ktiles), single N tile, reads act0 =========
  {
    const int m    = lane & 15;
    const int asel = (lane >> 4) * 8;
    v8f o = zero8();
#pragma unroll
    for (int kt = 0; kt < 8; ++kt) {
      Frag16 a;
      const uint32_t aoff = act0 + (uint32_t)(m * 512 + kt * 64 + asel * 2);
      a.f[0] = *(const float4*)(smem + aoff);
      a.f[1] = *(const float4*)(smem + aoff + 32);
      Frag16 b;
      const uint32_t boff = (uint32_t)(kt * 1024 + lane * 32);
      b.f[0] = *(const float4*)(smem + boff);
      b.f[1] = *(const float4*)(smem + boff + 16);
      o = __builtin_amdgcn_wmma_f32_16x16x32_f16(
          false, a.h, false, b.h, (short)0, o, false, false);
    }
    const int col = lane & 15;
    const int rb  = (lane >> 4) * 8;
    if (col < 3) {
      float bv = b4[col];
#pragma unroll
      for (int j = 0; j < 8; ++j)
        *(float*)(smem + OUT_OFF + (uint32_t)wv * 256 +
                  (uint32_t)((rb + j) * 16 + col * 4)) = o[j] + bv;
    }
  }
  __syncthreads();

  // ---- local-ensemble blend (areas diagonally swapped), lanes 0..11 per wave ----
  if (lane < 12) {
    const int qi  = lane / 3;
    const int col = lane - qi * 3;
    const uint32_t ab = AREA_OFF + (uint32_t)(wv * 16 + qi * 4) * 4;
    float a0 = *(const float*)(smem + ab + 0);
    float a1 = *(const float*)(smem + ab + 4);
    float a2 = *(const float*)(smem + ab + 8);
    float a3 = *(const float*)(smem + ab + 12);
    float tot = a0 + a1 + a2 + a3;
    const uint32_t ob = OUT_OFF + (uint32_t)wv * 256 + (uint32_t)(qi * 4) * 16 +
                        (uint32_t)col * 4;
    float p0 = *(const float*)(smem + ob + 0 * 16);
    float p1 = *(const float*)(smem + ob + 1 * 16);
    float p2 = *(const float*)(smem + ob + 2 * 16);
    float p3 = *(const float*)(smem + ob + 3 * 16);
    float res = (p0 * a3 + p1 * a2 + p2 * a1 + p3 * a0) / tot;
    const int bq = (base_row >> 2) + qi;
    out[(size_t)bq * 3 + col] = res;
  }
}

// ================= host launcher =================
extern "C" void kernel_launch(void* const* d_in, const int* in_sizes, int n_in,
                              void* d_out, int out_size, void* d_ws, size_t ws_size,
                              hipStream_t stream) {
  const float* feat = (const float*)d_in[0];
  const float* grid = (const float*)d_in[1];
  const float* cell = (const float*)d_in[2];
  const float* w0 = (const float*)d_in[3];  const float* b0 = (const float*)d_in[4];
  const float* w1 = (const float*)d_in[5];  const float* b1 = (const float*)d_in[6];
  const float* w2 = (const float*)d_in[7];  const float* b2 = (const float*)d_in[8];
  const float* w3 = (const float*)d_in[9];  const float* b3 = (const float*)d_in[10];
  const float* w4 = (const float*)d_in[11]; const float* b4 = (const float*)d_in[12];
  _Float16* wf = (_Float16*)d_ws;

  hipFuncSetAttribute((const void*)liif_main_kernel,
                      hipFuncAttributeMaxDynamicSharedMemorySize, SMEM_BYTES);

  // 1) pre-swizzle weights into WMMA B-fragment layout (f32 -> f16)
  liif_prep_kernel<<<(TOT_FRAGS * 32 + 255) / 256, 256, 0, stream>>>(
      w0, w1, w2, w3, w4, wf);

  // 2) fused gather + 5-layer MLP + ensemble blend
  //    262144 rows / (8 waves * 16 rows) = 2048 blocks
  liif_main_kernel<<<2048, 256, SMEM_BYTES, stream>>>(
      feat, grid, cell, b0, b1, b2, b3, b4, wf, (float*)d_out);
}
